// ContrastiveLoss_4415226380360
// MI455X (gfx1250) — matrix operand order repaired and checked by reading, not compiled, not run on hardware
//
#include <hip/hip_runtime.h>
#include <math.h>

// ---------------------------------------------------------------------------
// SimCLR NT-Xent loss, MI455X (gfx1250, wave32, WMMA).
// loss = -mean_r( sim[r, r^B2/2] - logsumexp_{c!=r}(sim[r,c]) ),
// sim = (Z Z^T) / T with Z row-normalized, T = 0.5.
// Gram matrix via v_wmma_f32_16x16x32_bf16 with hi/lo bf16 split of fp32
// (3 WMMAs per K=32 chunk ~ fp32 accuracy), fused flash-style online LSE.
// ---------------------------------------------------------------------------

typedef __attribute__((ext_vector_type(16))) __bf16         v16bf;
typedef __attribute__((ext_vector_type(8)))  float          v8f;
typedef __attribute__((ext_vector_type(8)))  unsigned short u16x8;
typedef __attribute__((ext_vector_type(16))) unsigned short u16x16;

static constexpr int   DIM      = 256;   // feature dim
static constexpr float TEMP_INV = 2.0f;  // 1 / TEMPERATURE

__device__ __forceinline__ unsigned short f32_to_bf16_rne(float f) {
  unsigned u = __float_as_uint(f);
  u += 0x7FFFu + ((u >> 16) & 1u);
  return (unsigned short)(u >> 16);
}
__device__ __forceinline__ float bf16_to_f32(unsigned short h) {
  return __uint_as_float(((unsigned)h) << 16);
}
__device__ __forceinline__ v16bf as_bf16x16(u16x16 u) {
  union { u16x16 u; v16bf b; } c;
  c.u = u;
  return c.b;
}

// ---------------------------------------------------------------------------
// Kernel 1: L2-normalize each of the 2B rows, split fp32 -> bf16 hi + lo.
// One wave per row; 8 elements per lane.
// ---------------------------------------------------------------------------
__global__ void nrm_split(const float* __restrict__ emb_i,
                          const float* __restrict__ emb_j,
                          unsigned short* __restrict__ zhi,
                          unsigned short* __restrict__ zlo, int Bn) {
  const int row = blockIdx.x;
  const int t   = threadIdx.x;  // 0..31
  const float* src = (row < Bn) ? (emb_i + (size_t)row * DIM)
                                : (emb_j + (size_t)(row - Bn) * DIM);
  float4 v0 = ((const float4*)src)[t * 2 + 0];
  float4 v1 = ((const float4*)src)[t * 2 + 1];
  float v[8] = {v0.x, v0.y, v0.z, v0.w, v1.x, v1.y, v1.z, v1.w};
  float ss = 0.f;
#pragma unroll
  for (int k = 0; k < 8; ++k) ss += v[k] * v[k];
#pragma unroll
  for (int off = 16; off >= 1; off >>= 1) ss += __shfl_xor(ss, off, 32);
  const float scale = 1.0f / fmaxf(sqrtf(ss), 1e-12f);
  const size_t base = (size_t)row * DIM + (size_t)t * 8;
#pragma unroll
  for (int k = 0; k < 8; ++k) {
    float z = v[k] * scale;
    unsigned short hi = f32_to_bf16_rne(z);
    zhi[base + k] = hi;
    zlo[base + k] = f32_to_bf16_rne(z - bf16_to_f32(hi));
  }
}

// ---------------------------------------------------------------------------
// Kernel 2: per-row  t_r = sim[r, partner] - logsumexp_{c != r} sim[r, c].
// Block = 8 waves, owns 16 rows (one WMMA M-panel, resident in registers).
// Wave w streams column tiles w, w+8, ... ; 3 bf16 WMMAs per K=32 chunk.
// ---------------------------------------------------------------------------
__global__ void __launch_bounds__(256) simclr_rows(
    const unsigned short* __restrict__ zhi,
    const unsigned short* __restrict__ zlo,
    float* __restrict__ rowvals, int R) {
  const int rowBase = blockIdx.x * 16;
  const int lane = threadIdx.x & 31;
  const int wave = threadIdx.x >> 5;  // 0..7
  const int half = lane >> 4;         // 0/1  (K-split per ISA layout)
  const int n    = lane & 15;         // A: row-in-tile, B: col-in-tile

  __shared__ float sm[8][16];
  __shared__ float ssum[8][16];
  __shared__ float spos[8][16];

  const float NEG_INF = -__builtin_inff();

  // --- resident A panel (16 rows x 256 K), hi+lo, contiguous K per lane half (same order as B) ---
  // lane L in [0,16): M = L, elems 0..15 = K c+0..15
  // lane L in [16,32): M = L-16, elems 0..15 = K c+16..31
  v16bf a_hi[8], a_lo[8];
  {
    const size_t arow = (size_t)(rowBase + n) * DIM;
#pragma unroll
    for (int kc = 0; kc < 8; ++kc) {
      const size_t off = arow + (size_t)kc * 32 + (size_t)half * 16;
      u16x8 x0 = *(const u16x8*)(zhi + off);
      u16x8 x1 = *(const u16x8*)(zhi + off + 8);
      a_hi[kc] = as_bf16x16(__builtin_shufflevector(
          x0, x1, 0, 1, 2, 3, 4, 5, 6, 7, 8, 9, 10, 11, 12, 13, 14, 15));
      u16x8 y0 = *(const u16x8*)(zlo + off);
      u16x8 y1 = *(const u16x8*)(zlo + off + 8);
      a_lo[kc] = as_bf16x16(__builtin_shufflevector(
          y0, y1, 0, 1, 2, 3, 4, 5, 6, 7, 8, 9, 10, 11, 12, 13, 14, 15));
    }
  }

  float m[8], s[8], p[8];
#pragma unroll
  for (int j = 0; j < 8; ++j) {
    m[j] = NEG_INF;
    s[j] = 0.0f;
    p[j] = NEG_INF;
  }

  const int nTiles  = R / 16;
  const int partnerX = R >> 1;  // 4096; partner col = row ^ partnerX

  for (int tile = wave; tile < nTiles; tile += 8) {
    const int colBase = tile * 16;
    // B layout: lane holds col N = n, elems 0..15 = K c+half*16 .. +15
    const size_t bbase = (size_t)(colBase + n) * DIM + (size_t)half * 16;

    if (tile + 8 < nTiles) {  // prefetch next B tile (global_prefetch)
      __builtin_prefetch(zhi + bbase + (size_t)128 * DIM, 0, 1);
      __builtin_prefetch(zlo + bbase + (size_t)128 * DIM, 0, 1);
    }

    v8f acc = {};
#pragma unroll
    for (int kc = 0; kc < 8; ++kc) {
      v16bf bhi = as_bf16x16(*(const u16x16*)(zhi + bbase + kc * 32));
      v16bf blo = as_bf16x16(*(const u16x16*)(zlo + bbase + kc * 32));
      acc = __builtin_amdgcn_wmma_f32_16x16x32_bf16(false, a_hi[kc], false,
                                                    bhi, (short)0, acc,
                                                    false, false);
      acc = __builtin_amdgcn_wmma_f32_16x16x32_bf16(false, a_hi[kc], false,
                                                    blo, (short)0, acc,
                                                    false, false);
      acc = __builtin_amdgcn_wmma_f32_16x16x32_bf16(false, a_lo[kc], false,
                                                    bhi, (short)0, acc,
                                                    false, false);
    }

    // C/D layout: VGPR j, lane -> element (M = j + 8*half, N = n)
    const int col = colBase + n;
#pragma unroll
    for (int j = 0; j < 8; ++j) {
      const int row = rowBase + j + half * 8;
      if (col != row) {
        const float v = acc[j] * TEMP_INV;
        if (col == (row ^ partnerX)) p[j] = fmaxf(p[j], v);
        if (v > m[j]) {
          s[j] = s[j] * __expf(m[j] - v) + 1.0f;
          m[j] = v;
        } else {
          s[j] += __expf(v - m[j]);
        }
      }
    }
  }

  // --- combine the 16 column-lanes of each half (same row, same VGPR j) ---
#pragma unroll
  for (int j = 0; j < 8; ++j) {
#pragma unroll
    for (int mask = 1; mask <= 8; mask <<= 1) {
      float om = __shfl_xor(m[j], mask, 32);
      float os = __shfl_xor(s[j], mask, 32);
      float op = __shfl_xor(p[j], mask, 32);
      p[j] = fmaxf(p[j], op);
      if (om > m[j]) {
        s[j] = s[j] * __expf(m[j] - om) + os;
        m[j] = om;
      } else {
        s[j] += os * __expf(om - m[j]);
      }
    }
  }
  if (n == 0) {
#pragma unroll
    for (int j = 0; j < 8; ++j) {
      sm[wave][half * 8 + j]   = m[j];
      ssum[wave][half * 8 + j] = s[j];
      spos[wave][half * 8 + j] = p[j];
    }
  }
  __syncthreads();

  // --- combine the 8 waves, emit t_r ---
  if (threadIdx.x < 16) {
    const int r = threadIdx.x;
    float M = NEG_INF, S = 0.0f, P = NEG_INF;
#pragma unroll
    for (int w = 0; w < 8; ++w) {
      const float mw = sm[w][r];
      const float sw = ssum[w][r];
      P = fmaxf(P, spos[w][r]);
      if (mw > M) {
        S = S * __expf(M - mw) + sw;
        M = mw;
      } else {
        S += sw * __expf(mw - M);
      }
    }
    rowvals[rowBase + r] = P - (M + __logf(S));
  }
}

// ---------------------------------------------------------------------------
// Kernel 3: loss = -mean(rowvals)
// ---------------------------------------------------------------------------
__global__ void reduce_loss(const float* __restrict__ rowvals,
                            float* __restrict__ out, int R) {
  __shared__ float sh[256];
  float acc = 0.0f;
  for (int i = threadIdx.x; i < R; i += 256) acc += rowvals[i];
  sh[threadIdx.x] = acc;
  __syncthreads();
  for (int off = 128; off >= 1; off >>= 1) {
    if ((int)threadIdx.x < off) sh[threadIdx.x] += sh[threadIdx.x + off];
    __syncthreads();
  }
  if (threadIdx.x == 0) out[0] = -(sh[0] / (float)R);
}

// ---------------------------------------------------------------------------
extern "C" void kernel_launch(void* const* d_in, const int* in_sizes, int n_in,
                              void* d_out, int out_size, void* d_ws,
                              size_t ws_size, hipStream_t stream) {
  const float* emb_i = (const float*)d_in[0];
  const float* emb_j = (const float*)d_in[1];
  const int Bn = in_sizes[0] / DIM;  // 4096
  const int R  = 2 * Bn;             // 8192 rows total

  // workspace layout: bf16 hi plane | bf16 lo plane | per-row results
  unsigned short* zhi = (unsigned short*)d_ws;
  unsigned short* zlo = zhi + (size_t)R * DIM;
  float* rowvals      = (float*)(zlo + (size_t)R * DIM);

  nrm_split<<<R, 32, 0, stream>>>(emb_i, emb_j, zhi, zlo, Bn);
  simclr_rows<<<R / 16, 256, 0, stream>>>(zhi, zlo, rowvals, R);
  reduce_loss<<<1, 256, 0, stream>>>(rowvals, (float*)d_out, R);
}
